// RNNLayer_10977936408826
// MI455X (gfx1250) — matrix operand order repaired and checked by reading, not compiled, not run on hardware
//
#include <hip/hip_runtime.h>
#include <hip/hip_bf16.h>
#include <math.h>

// Problem constants (match reference)
#define B_GRAPHS 64
#define N_PER    2048
#define T_TOTAL  (B_GRAPHS * N_PER)   // 131072
#define D_HID    256
#define U_DIM    128
#define U3       384
#define S_LEN    N_PER                // 2048

typedef __attribute__((ext_vector_type(16))) _Float16 v16h;
typedef __attribute__((ext_vector_type(8)))  float    v8f;
typedef __attribute__((ext_vector_type(4)))  unsigned int u32x4;
typedef __attribute__((ext_vector_type(8)))  int      i32x8;
typedef __attribute__((ext_vector_type(4)))  int      i32x4;

#if __has_builtin(__builtin_amdgcn_tensor_load_to_lds) && \
    __has_builtin(__builtin_amdgcn_s_wait_tensorcnt)
#define HAVE_TDM 1
#else
#define HAVE_TDM 0
#endif

static __device__ __forceinline__ int lane_id() { return threadIdx.x & 31; }

// ---- WMMA fragment loaders (ISA 7.12.2 wave32 layouts) -------------------

// A fragment (16x32 f16) from f32 row-major [rows x ld]; rows m0..m0+15, k-tile kt
static __device__ __forceinline__ v16h a_frag_f32(const float* __restrict__ base,
                                                  int ld, int m0, int kt) {
  const int l   = lane_id();
  const int row = m0 + (l & 15);
  const int kb  = kt * 32 + ((l >> 4) & 1) * 8;   // lanes16-31 take the +8 K-half
  const float* p = base + (size_t)row * ld + kb;
  v16h a;
#pragma unroll
  for (int j = 0; j < 8; ++j) a[j] = (_Float16)p[j];          // K = kb .. kb+7
#pragma unroll
  for (int j = 0; j < 8; ++j) a[8 + j] = (_Float16)p[16 + j]; // K = kb+16 .. kb+23
  return a;
}

// A fragment from LDS f16 row-major [16 x ld]
static __device__ __forceinline__ v16h a_frag_lds(const _Float16* __restrict__ base,
                                                  int ld, int kt) {
  const int l   = lane_id();
  const int row = l & 15;
  const int kb  = kt * 32 + ((l >> 4) & 1) * 8;
  const _Float16* p = base + row * ld + kb;
  v16h a;
#pragma unroll
  for (int j = 0; j < 8; ++j) a[j] = p[j];
#pragma unroll
  for (int j = 0; j < 8; ++j) a[8 + j] = p[16 + j];
  return a;
}

// B fragment (32x16 f16) from N-major transposed weights in LDS: Wt[n*ldk + k]
static __device__ __forceinline__ v16h b_frag_lds(const _Float16* __restrict__ Wt,
                                                  int ldk, int n0, int kt) {
  const int l  = lane_id();
  const int n  = n0 + (l & 15);
  const int kb = kt * 32 + ((l >> 4) & 1) * 8;    // lanes16-31 take the +8 K-half
  const _Float16* p = Wt + n * ldk + kb;
  v16h b;
#pragma unroll
  for (int j = 0; j < 8; ++j) b[j] = p[j];
#pragma unroll
  for (int j = 0; j < 8; ++j) b[8 + j] = p[16 + j];
  return b;
}

// D tile (16x16 f32) store to row-major [ .. x ldn ]
static __device__ __forceinline__ void store_d_global(float* __restrict__ out, int ldn,
                                                      int m0, int n0, v8f acc) {
  const int l     = lane_id();
  const int col   = n0 + (l & 15);
  const int rbase = m0 + ((l >> 4) & 1) * 8;
#pragma unroll
  for (int i = 0; i < 8; ++i) out[(size_t)(rbase + i) * ldn + col] = acc[i];
}

#if HAVE_TDM
// ---- Tensor Data Mover: DMA a [16 x 384] f32 tile (row stride S*384 elems)
// from global into contiguous LDS.  D# per ISA 08_async_tensor §8.3/8.4.
static __device__ __forceinline__ void tdm_load_tile(const float* gsrc, void* ldst) {
  const unsigned long long ga = (unsigned long long)gsrc;
  const unsigned lds = (unsigned)(unsigned long long)ldst; // low 32b = LDS offset
  u32x4 g0;
  g0[0] = 1u;                                       // count=1 (valid), no gather
  g0[1] = lds;                                      // lds_addr
  g0[2] = (unsigned)(ga & 0xFFFFFFFFull);           // global_addr[31:0]
  g0[3] = (unsigned)((ga >> 32) & 0x1FFFFFFull)     // global_addr[56:32]
        | (2u << 30);                               // type = 2 (image)
  i32x8 g1;
  g1[0] = 0x00020000;                               // wg_mask=0, data_size=4B
  g1[1] = (int)(U3 << 16);                          // tensor_dim0 = 384
  g1[2] = (int)(16 << 16);                          // tensor_dim1 = 16
  g1[3] = (int)(U3 << 16);                          // tile_dim0 = 384
  g1[4] = 16;                                       // tile_dim1 = 16, tile_dim2 = 0
  g1[5] = (int)(S_LEN * U3);                        // tensor_dim0_stride = 786432
  g1[6] = 0;
  g1[7] = 0;
  i32x4 z4 = {0, 0, 0, 0};                          // groups 2/3 unused (2D)
#if defined(__clang_major__) && (__clang_major__ >= 23)
  i32x8 z8 = {0, 0, 0, 0, 0, 0, 0, 0};
  __builtin_amdgcn_tensor_load_to_lds(g0, g1, z4, z4, z8, 0);
#else
  __builtin_amdgcn_tensor_load_to_lds(g0, g1, z4, z4, 0);
#endif
}
#endif

// ---- Utility kernels -----------------------------------------------------

__global__ void zero_f32_kernel(float* __restrict__ p, size_t n) {
  size_t i = (size_t)blockIdx.x * blockDim.x + threadIdx.x;
  const size_t stride = (size_t)gridDim.x * blockDim.x;
  for (; i < n; i += stride) p[i] = 0.0f;
}

// Pack leaves into dense [B,S,D] + mask. One block (64 threads) per node.
__global__ void pack_kernel(const float* __restrict__ states,
                            const int* __restrict__ pos,
                            float* __restrict__ seq, int* __restrict__ mask) {
  const int t = blockIdx.x;
  const int p = pos[t];
  if (p == 0) return;                         // non-leaf: slot stays zero/masked
  const int g    = t >> 11;                   // N_PER == 2048 (uniform graphs)
  const int slot = p - 1;
  const size_t row = (size_t)g * S_LEN + slot;
  const float4* s4 = (const float4*)(states + (size_t)t * D_HID);
  float4*       d4 = (float4*)(seq + row * D_HID);
  d4[threadIdx.x] = s4[threadIdx.x];          // 64 threads x float4 = 256 floats
  if (threadIdx.x == 0) mask[row] = 1;
}

// ---- Input-projection GEMM: XP[M,384] = f16wmma( X[M,256] @ K[256,384] ) + b0
// Block: 256 threads (8 waves), 128 rows per block; weights transposed to LDS.
__global__ void __launch_bounds__(256)
xproj_gemm_kernel(const float* __restrict__ X,   // [B*S, 256] f32
                  const float* __restrict__ K,   // [256, 384] f32 row-major
                  const float* __restrict__ b0,  // [384]
                  float* __restrict__ XP) {      // [B*S, 384] f32
  extern __shared__ char smem[];
  _Float16* Kt = (_Float16*)smem;                // [384][256] f16 = 192KB
  for (int idx = threadIdx.x; idx < 256 * 384; idx += blockDim.x) {
    const int k = idx / U3, n = idx % U3;        // coalesced global read
    Kt[n * 256 + k] = (_Float16)K[idx];
  }
  __syncthreads();

  const int w  = threadIdx.x >> 5;
  const int m0 = blockIdx.x * 128 + w * 16;      // this wave's 16 rows
  const int l  = lane_id();

  // Preload per-lane bias for all 24 n-tiles (keeps the hot loop global-free)
  float biasv[24];
#pragma unroll
  for (int nt = 0; nt < 24; ++nt) biasv[nt] = b0[nt * 16 + (l & 15)];

  // Cache A fragments for the full K=256 depth (8 k-tiles)
  v16h a[8];
#pragma unroll
  for (int kt = 0; kt < 8; ++kt) a[kt] = a_frag_f32(X, D_HID, m0, kt);

  for (int nt = 0; nt < 24; ++nt) {
    // Fetch the whole 256-deep B column strip first, then run WMMAs
    // back-to-back (one DScnt drain per tile instead of one per WMMA).
    v16h b[8];
#pragma unroll
    for (int kt = 0; kt < 8; ++kt) b[kt] = b_frag_lds(Kt, 256, nt * 16, kt);
    v8f acc;
#pragma unroll
    for (int i = 0; i < 8; ++i) acc[i] = biasv[nt];
#pragma unroll
    for (int kt = 0; kt < 8; ++kt)
      acc = __builtin_amdgcn_wmma_f32_16x16x32_f16(false, a[kt], false, b[kt],
                                                   (short)0, acc, false, false);
    store_d_global(XP, U3, m0, nt * 16, acc);
  }
}

// ---- Persistent bidirectional GRU scan -----------------------------------
// Grid = 8 blocks: dir = blockIdx>>2 (0 fwd / 1 bwd), bg = blockIdx&3 (16 batches).
// R^T lives in LDS as f16; the 12 B-fragments stay VGPR-resident for all 2048
// steps; the next step's x-projection tile is DMA'd into an LDS double buffer
// by the Tensor Data Mover while the current step's WMMAs run.
__global__ void __launch_bounds__(256)
gru_scan_kernel(const float* __restrict__ XPf, const float* __restrict__ XPb,
                const float* __restrict__ Rf,  const float* __restrict__ Rb,
                const float* __restrict__ b1f, const float* __restrict__ b1b,
                const int* __restrict__ mask, float* __restrict__ Y) {
  const int dir = blockIdx.x >> 2;
  const int bg  = blockIdx.x & 3;
  const float* XP = dir ? XPb : XPf;
  const float* R  = dir ? Rb  : Rf;
  const float* b1 = dir ? b1b : b1f;
  const int dirOff = dir ? U_DIM : 0;

  extern __shared__ char smem[];
  _Float16* Rt   = (_Float16*)smem;                             // [384][128]  96KB
  float*    hp   = (float*)(smem + 98304);                      // [16][384]   24KB
  float*    hf   = (float*)(smem + 98304 + 24576);              // [16][128]    8KB
  _Float16* h16  = (_Float16*)(smem + 98304 + 24576 + 8192);    // [16][128]    4KB
  float*    xb0  = (float*)(smem + 135168);                     // [16][384]   24KB
  float*    xb1  = (float*)(smem + 135168 + 24576);             // [16][384]   24KB

  for (int idx = threadIdx.x; idx < U_DIM * U3; idx += blockDim.x) {
    const int k = idx / U3, n = idx % U3;
    Rt[n * U_DIM + k] = (_Float16)R[idx];
  }
  for (int idx = threadIdx.x; idx < 16 * U_DIM; idx += blockDim.x) {
    hf[idx] = 0.0f; h16[idx] = (_Float16)0.0f;
  }

  const int w = threadIdx.x >> 5;
  const int l = lane_id();
  const int ntBase = w * 3;                      // 8 waves x 3 n-tiles = 24
  const size_t batchRow0 = (size_t)(bg * 16) * S_LEN;

#if HAVE_TDM
  // Prologue: DMA step 0's projection tile into buffer 0, drain TENSORcnt.
  {
    const int t0 = dir ? (S_LEN - 1) : 0;
    if (threadIdx.x == 0) {
      tdm_load_tile(XP + (batchRow0 + t0) * U3, xb0);
      __builtin_amdgcn_s_wait_tensorcnt(0);
    }
  }
#endif
  __syncthreads();

  // Weight fragments resident in VGPRs across the whole scan (12 x 8 dwords)
  v16h Bfr[3][4];
#pragma unroll
  for (int i = 0; i < 3; ++i)
#pragma unroll
    for (int kt = 0; kt < 4; ++kt)
      Bfr[i][kt] = b_frag_lds(Rt, U_DIM, (ntBase + i) * 16, kt);

  float biasv[3];
#pragma unroll
  for (int i = 0; i < 3; ++i) biasv[i] = b1[(ntBase + i) * 16 + (l & 15)];

  for (int s = 0; s < S_LEN; ++s) {
    const int t = dir ? (S_LEN - 1 - s) : s;

#if HAVE_TDM
    // Kick the DMA for step s+1 into the other buffer; it overlaps the WMMAs.
    if (s + 1 < S_LEN && threadIdx.x == 0) {
      const int tn = dir ? (t - 1) : (t + 1);
      tdm_load_tile(XP + (batchRow0 + tn) * U3, (s & 1) ? xb0 : xb1);
    }
#endif

    // hp = h @ R + b1 : M=16 batches, N=384, K=128  -> 12 WMMAs / wave
    v8f acc[3];
#pragma unroll
    for (int i = 0; i < 3; ++i)
#pragma unroll
      for (int e = 0; e < 8; ++e) acc[i][e] = biasv[i];
#pragma unroll
    for (int kt = 0; kt < 4; ++kt) {
      v16h a = a_frag_lds(h16, U_DIM, kt);
#pragma unroll
      for (int i = 0; i < 3; ++i)
        acc[i] = __builtin_amdgcn_wmma_f32_16x16x32_f16(false, a, false, Bfr[i][kt],
                                                        (short)0, acc[i], false, false);
    }
#pragma unroll
    for (int i = 0; i < 3; ++i) {
      const int col = (ntBase + i) * 16 + (l & 15);
      const int rb  = ((l >> 4) & 1) * 8;
#pragma unroll
      for (int e = 0; e < 8; ++e) hp[(rb + e) * U3 + col] = acc[i][e];
    }
    __syncthreads();

    // Gate math: 16*128 elements over 256 threads (8 each)
#if HAVE_TDM
    const float* xtile = (s & 1) ? xb1 : xb0;     // this step's DMA'd tile
#endif
    for (int e = threadIdx.x; e < 16 * U_DIM; e += blockDim.x) {
      const int bl = e >> 7;
      const int u  = e & (U_DIM - 1);
      const int gb = bg * 16 + bl;
      const size_t row = (size_t)gb * S_LEN + t;
#if HAVE_TDM
      const float* xp = xtile + bl * U3;          // LDS (prefetched by TDM)
#else
      const float* xp = XP + row * U3;            // global fallback
#endif
      const float xz = xp[u], xr = xp[U_DIM + u], xh = xp[2 * U_DIM + u];
      const float hz = hp[bl * U3 + u];
      const float hr = hp[bl * U3 + U_DIM + u];
      const float hc = hp[bl * U3 + 2 * U_DIM + u];
      const float z    = 1.0f / (1.0f + __expf(-(xz + hz)));
      const float r    = 1.0f / (1.0f + __expf(-(xr + hr)));
      const float cand = tanhf(xh + r * hc);
      const float hprev = hf[e];
      const float hn    = z * hprev + (1.0f - z) * cand;
      const float hcur  = mask[row] ? hn : hprev;   // padded steps carry state
      hf[e]  = hcur;
      h16[e] = (_Float16)hcur;
      Y[row * D_HID + dirOff + u] = hcur;
#if !HAVE_TDM
      if ((s + 1 < S_LEN) && ((u & 15) == 0)) {
        const int tn = dir ? (t - 1) : (t + 1);
        const float* np = XP + ((size_t)gb * S_LEN + tn) * U3;
        __builtin_prefetch(np + u, 0, 1);
        __builtin_prefetch(np + U_DIM + u, 0, 1);
        __builtin_prefetch(np + 2 * U_DIM + u, 0, 1);
      }
#endif
    }
#if HAVE_TDM
    // Make sure next step's tile has landed before everyone passes the barrier.
    if (s + 1 < S_LEN && threadIdx.x == 0)
      __builtin_amdgcn_s_wait_tensorcnt(0);
#endif
    __syncthreads();
  }
}

// Scatter back: leaves take RNN output, non-leaves keep original state.
__global__ void unpack_kernel(const float* __restrict__ states,
                              const int* __restrict__ pos,
                              const float* __restrict__ Y,
                              float* __restrict__ out) {
  const int t = blockIdx.x;
  const int p = pos[t];
  const int g = t >> 11;
  const float4* src = (p != 0)
      ? (const float4*)(Y + ((size_t)g * S_LEN + (p - 1)) * D_HID)
      : (const float4*)(states + (size_t)t * D_HID);
  float4* dst = (float4*)(out + (size_t)t * D_HID);
  dst[threadIdx.x] = src[threadIdx.x];
}

// ---- Host driver ---------------------------------------------------------

extern "C" void kernel_launch(void* const* d_in, const int* in_sizes, int n_in,
                              void* d_out, int out_size, void* d_ws, size_t ws_size,
                              hipStream_t stream) {
  (void)in_sizes; (void)n_in; (void)out_size; (void)ws_size;
  const float* states = (const float*)d_in[0];
  const int*   pos    = (const int*)d_in[1];
  // d_in[2] graph_sizes (uniform N_PER), d_in[3] training (== 0): unused
  const float* K_fwd  = (const float*)d_in[4];  // [2,256,384]
  const float* R_fwd  = (const float*)d_in[5];  // [2,128,384]
  const float* b_fwd  = (const float*)d_in[6];  // [2,2,384]
  const float* K_bwd  = (const float*)d_in[7];
  const float* R_bwd  = (const float*)d_in[8];
  const float* b_bwd  = (const float*)d_in[9];

  const size_t rowsD  = (size_t)T_TOTAL * D_HID;   // elements of a [B,S,256] buf
  const size_t rows3U = (size_t)T_TOTAL * U3;      // elements of a [B,S,384] buf

  char* ws = (char*)d_ws;
  float* seq  = (float*)ws;                         // [B,S,256]; reused as y1
  float* y0   = (float*)(ws + rowsD * 4);           // [B,S,256]
  float* xpF  = (float*)(ws + 2 * rowsD * 4);       // [B,S,384]
  float* xpB  = xpF + rows3U;                       // [B,S,384]
  int*   mask = (int*)(xpB + rows3U);               // [B,S]
  float* y1   = seq;

  const size_t gemmLds = (size_t)U3 * D_HID * sizeof(_Float16);        // 192KB
  const size_t scanLds = 98304 + 24576 + 8192 + 4096 + 2 * 24576;      // 184320B

  // Zero the packed sequence and mask (ws is poisoned by the harness)
  zero_f32_kernel<<<8192, 256, 0, stream>>>(seq, rowsD);
  zero_f32_kernel<<<64, 256, 0, stream>>>((float*)mask, (size_t)B_GRAPHS * S_LEN);

  pack_kernel<<<T_TOTAL, 64, 0, stream>>>(states, pos, seq, mask);

  // ---- Layer 0 ----
  xproj_gemm_kernel<<<T_TOTAL / 128, 256, gemmLds, stream>>>(seq, K_fwd, b_fwd, xpF);
  xproj_gemm_kernel<<<T_TOTAL / 128, 256, gemmLds, stream>>>(seq, K_bwd, b_bwd, xpB);
  gru_scan_kernel<<<8, 256, scanLds, stream>>>(xpF, xpB, R_fwd, R_bwd,
                                               b_fwd + U3, b_bwd + U3, mask, y0);

  // ---- Layer 1 ----
  const float* K_f1 = K_fwd + (size_t)D_HID * U3;
  const float* K_b1 = K_bwd + (size_t)D_HID * U3;
  const float* R_f1 = R_fwd + (size_t)U_DIM * U3;
  const float* R_b1 = R_bwd + (size_t)U_DIM * U3;
  const float* bF1  = b_fwd + 2 * U3;
  const float* bB1  = b_bwd + 2 * U3;
  xproj_gemm_kernel<<<T_TOTAL / 128, 256, gemmLds, stream>>>(y0, K_f1, bF1, xpF);
  xproj_gemm_kernel<<<T_TOTAL / 128, 256, gemmLds, stream>>>(y0, K_b1, bB1, xpB);
  gru_scan_kernel<<<8, 256, scanLds, stream>>>(xpF, xpB, R_f1, R_b1,
                                               bF1 + U3, bB1 + U3, mask, y1);

  unpack_kernel<<<T_TOTAL, 64, 0, stream>>>(states, pos, y1, (float*)d_out);
}
